// MambaMixer_90460601189072
// MI455X (gfx1250) — matrix operand order repaired and checked by reading, not compiled, not run on hardware
//
#include <hip/hip_runtime.h>
#include <hip/hip_bf16.h>
#include <cstdint>

// ---------------- problem dims ----------------
constexpr int B_      = 2;
constexpr int L_      = 1024;
constexpr int DM      = 1024;          // d_model
constexpr int DI      = 2048;          // d_inner
constexpr int DS      = 16;            // d_state
constexpr int DC      = 4;             // d_conv
constexpr int DR      = 64;            // dt_rank
constexpr int NR      = B_ * L_;       // flattened rows = 2048
constexpr int DBLW    = DR + 2 * DS;   // 96

// ---------------- WMMA types ----------------
typedef __attribute__((ext_vector_type(16))) __bf16 v16bf;
typedef __attribute__((ext_vector_type(8)))  float  v8f;

union FragBF { v16bf v; uint32_t u[8]; };

// ---------------- GEMM: C[M,N] = A[M,K] * W[N,K]^T (bf16 in, f32 out) ------
// EPI: 0 = plain store, 1 = softplus(acc + bias[n])
constexpr int BM = 128, BN = 128, BK = 32;
constexpr int LSTR = 20;   // u32 stride per LDS row (16 data + 4 pad, keeps 16B align)

template<int EPI>
__global__ __launch_bounds__(256)
void gemm_bf16_kernel(const __hip_bfloat16* __restrict__ A,
                      const __hip_bfloat16* __restrict__ W,
                      const float* __restrict__ bias,
                      float* __restrict__ C,
                      int M, int N, int K)
{
    __shared__ alignas(16) uint32_t sA[BM * LSTR];
    __shared__ alignas(16) uint32_t sW[BN * LSTR];

    const int tid   = threadIdx.x;
    const int lane  = tid & 31;
    const int wave  = tid >> 5;
    const int wm    = wave & 3;     // wave M index (0..3) -> 32 rows each
    const int wn    = wave >> 2;    // wave N index (0..1) -> 64 cols each
    const int tileM = blockIdx.x * BM;
    const int tileN = blockIdx.y * BN;

    v8f zero = {0.f,0.f,0.f,0.f,0.f,0.f,0.f,0.f};
    v8f acc[2][4];
    #pragma unroll
    for (int ms = 0; ms < 2; ++ms)
        #pragma unroll
        for (int ns = 0; ns < 4; ++ns) acc[ms][ns] = zero;

    const int quad = tid & 3;    // which 16B chunk of a 64B (BK) row
    const int rrow = tid >> 2;   // 0..63
    const int hf   = (lane >> 4) & 1;
    const int lm   = lane & 15;

    // per-lane LDS byte addresses for async global->LDS staging
    const uint32_t ldsA0 = (uint32_t)(uintptr_t)&sA[(rrow     ) * LSTR + quad * 4];
    const uint32_t ldsA1 = (uint32_t)(uintptr_t)&sA[(rrow + 64) * LSTR + quad * 4];
    const uint32_t ldsW0 = (uint32_t)(uintptr_t)&sW[(rrow     ) * LSTR + quad * 4];
    const uint32_t ldsW1 = (uint32_t)(uintptr_t)&sW[(rrow + 64) * LSTR + quad * 4];

    for (int k0 = 0; k0 < K; k0 += BK) {
        // prefetch next A stripe (global_prefetch_b8)
        if (k0 + BK < K)
            __builtin_prefetch(A + (size_t)(tileM + rrow) * K + (k0 + BK), 0, 0);

        // ---- async global -> LDS staging (ASYNCcnt path), 16B per lane ----
        {
            uint64_t gA0 = (uint64_t)(uintptr_t)(A + (size_t)(tileM + rrow     ) * K + k0 + quad * 8);
            uint64_t gA1 = (uint64_t)(uintptr_t)(A + (size_t)(tileM + rrow + 64) * K + k0 + quad * 8);
            asm volatile("global_load_async_to_lds_b128 %0, %1, off"
                         :: "v"(ldsA0), "v"(gA0) : "memory");
            asm volatile("global_load_async_to_lds_b128 %0, %1, off"
                         :: "v"(ldsA1), "v"(gA1) : "memory");
            int gn0 = tileN + rrow;
            int gn1 = tileN + rrow + 64;
            if (gn0 < N) {
                uint64_t gW0 = (uint64_t)(uintptr_t)(W + (size_t)gn0 * K + k0 + quad * 8);
                asm volatile("global_load_async_to_lds_b128 %0, %1, off"
                             :: "v"(ldsW0), "v"(gW0) : "memory");
            } else {
                uint4 z4 = {0u,0u,0u,0u};
                *(uint4*)&sW[(rrow     ) * LSTR + quad * 4] = z4;
            }
            if (gn1 < N) {
                uint64_t gW1 = (uint64_t)(uintptr_t)(W + (size_t)gn1 * K + k0 + quad * 8);
                asm volatile("global_load_async_to_lds_b128 %0, %1, off"
                             :: "v"(ldsW1), "v"(gW1) : "memory");
            } else {
                uint4 z4 = {0u,0u,0u,0u};
                *(uint4*)&sW[(rrow + 64) * LSTR + quad * 4] = z4;
            }
            asm volatile("s_wait_asynccnt 0x0" ::: "memory");
        }
        __syncthreads();

        // ---- gather fragments (ISA 7.12.2 16-bit layouts) ----
        FragBF fa[2], fb[4];
        #pragma unroll
        for (int ms = 0; ms < 2; ++ms) {
            const uint32_t* src = &sA[(wm * 32 + ms * 16 + lm) * LSTR];
            const int o = hf * 4;   // K pairs {0..7,16..23} or {8..15,24..31}
            fa[ms].u[0] = src[0 + o];  fa[ms].u[1] = src[1 + o];
            fa[ms].u[2] = src[2 + o];  fa[ms].u[3] = src[3 + o];
            fa[ms].u[4] = src[8 + o];  fa[ms].u[5] = src[9 + o];
            fa[ms].u[6] = src[10 + o]; fa[ms].u[7] = src[11 + o];
        }
        #pragma unroll
        for (int ns = 0; ns < 4; ++ns) {
            const uint32_t* src = &sW[(wn * 64 + ns * 16 + lm) * LSTR + hf * 4];
            #pragma unroll
            for (int j = 0; j < 8; ++j) fb[ns].u[j] = src[(j & 3) + ((j >> 2) << 3)];
        }

        #pragma unroll
        for (int ms = 0; ms < 2; ++ms)
            #pragma unroll
            for (int ns = 0; ns < 4; ++ns)
                acc[ms][ns] = __builtin_amdgcn_wmma_f32_16x16x32_bf16(
                    false, fa[ms].v, false, fb[ns].v,
                    (short)0, acc[ms][ns], false, false);

        __syncthreads();
    }

    // ---- store D (C/D layout: vgpr r -> M=r (+8 hi lanes), N=lane&15) ----
    #pragma unroll
    for (int ms = 0; ms < 2; ++ms) {
        #pragma unroll
        for (int ns = 0; ns < 4; ++ns) {
            #pragma unroll
            for (int r = 0; r < 8; ++r) {
                int m = tileM + wm * 32 + ms * 16 + r + hf * 8;
                int n = tileN + wn * 64 + ns * 16 + lm;
                if (m < M && n < N) {
                    float v = acc[ms][ns][r];
                    if (EPI == 1) {
                        v += bias[n];
                        v = (v > 20.f) ? v : log1pf(__expf(v));   // softplus
                    }
                    C[(size_t)m * N + n] = v;
                }
            }
        }
    }
}

// ---------------- f32 -> bf16 conversion ----------------
__global__ void cvt_bf16_kernel(const float* __restrict__ src,
                                __hip_bfloat16* __restrict__ dst, int n)
{
    int i = blockIdx.x * blockDim.x + threadIdx.x;
    if (i < n) dst[i] = __float2bfloat16(src[i]);
}

// ---------------- causal depthwise conv (k=4) + SiLU ----------------
__global__ void conv_silu_kernel(const float* __restrict__ xz,  // NR x 2*DI
                                 const float* __restrict__ cw,  // DI x 4
                                 const float* __restrict__ cb,  // DI
                                 float* __restrict__ xc,
                                 __hip_bfloat16* __restrict__ xc_bf)
{
    int idx = blockIdx.x * blockDim.x + threadIdx.x;
    if (idx >= NR * DI) return;
    int e   = idx % DI;
    int row = idx / DI;          // b*L + l
    int l   = row % L_;
    float s = cb[e];
    #pragma unroll
    for (int k = 0; k < DC; ++k) {
        int lt = l + k - (DC - 1);
        if (lt >= 0)
            s += cw[e * DC + k] * xz[(size_t)(row + k - (DC - 1)) * (2 * DI) + e];
    }
    float r = s / (1.f + __expf(-s));      // silu
    xc[idx]    = r;
    xc_bf[idx] = __float2bfloat16(r);
}

// ---------------- slice dbl[:, :64] -> bf16 ----------------
__global__ void dtin_slice_kernel(const float* __restrict__ dbl,
                                  __hip_bfloat16* __restrict__ dtin)
{
    int idx = blockIdx.x * blockDim.x + threadIdx.x;
    if (idx >= NR * DR) return;
    int row = idx / DR, r = idx % DR;
    dtin[idx] = __float2bfloat16(dbl[(size_t)row * DBLW + r]);
}

// ---------------- selective scan: 16 lanes = 16 states of one channel -----
__global__ __launch_bounds__(256)
void scan_kernel(const float* __restrict__ dt,    // NR x DI (post-softplus)
                 const float* __restrict__ dbl,   // NR x 96 (B: 64..79, C: 80..95)
                 const float* __restrict__ xc,    // NR x DI
                 const float* __restrict__ A_log, // DI x 16
                 float* __restrict__ ys)          // NR x DI
{
    int tid = threadIdx.x;
    int n   = tid & 15;                              // state index
    int grp = (blockIdx.x * blockDim.x + tid) >> 4;  // channel 0..B*DI-1
    int b   = grp >> 11;                             // /2048
    int e   = grp & (DI - 1);

    const float Aen = -__expf(A_log[e * DS + n]);
    float h = 0.f;
    const float* dt_be = dt  + (size_t)b * L_ * DI + e;
    const float* xc_be = xc  + (size_t)b * L_ * DI + e;
    const float* dbl_b = dbl + (size_t)b * L_ * DBLW;
    float*       y_be  = ys  + (size_t)b * L_ * DI + e;

    for (int t = 0; t < L_; ++t) {
        float dtv = dt_be[(size_t)t * DI];          // broadcast within group
        float xv  = xc_be[(size_t)t * DI];
        float Bv  = dbl_b[(size_t)t * DBLW + DR + n];
        float Cv  = dbl_b[(size_t)t * DBLW + DR + DS + n];
        float dA  = __expf(dtv * Aen);
        h = h * dA + dtv * Bv * xv;
        float y = h * Cv;
        y += __shfl_xor(y, 1, 16);
        y += __shfl_xor(y, 2, 16);
        y += __shfl_xor(y, 4, 16);
        y += __shfl_xor(y, 8, 16);
        if (n == 0) y_be[(size_t)t * DI] = y;
    }
}

// ---------------- gate: y = (ys + xc*D) * silu(z), emit bf16 ----------------
__global__ void gate_kernel(const float* __restrict__ ys,
                            const float* __restrict__ xc,
                            const float* __restrict__ Dv,
                            const float* __restrict__ xz,   // z at cols [DI, 2DI)
                            __hip_bfloat16* __restrict__ y_bf)
{
    int idx = blockIdx.x * blockDim.x + threadIdx.x;
    if (idx >= NR * DI) return;
    int e = idx % DI, row = idx / DI;
    float z  = xz[(size_t)row * (2 * DI) + DI + e];
    float zs = z / (1.f + __expf(-z));
    float v  = (ys[idx] + xc[idx] * Dv[e]) * zs;
    y_bf[idx] = __float2bfloat16(v);
}

// ---------------- launcher ----------------
static inline size_t align256(size_t x) { return (x + 255) & ~size_t(255); }

extern "C" void kernel_launch(void* const* d_in, const int* in_sizes, int n_in,
                              void* d_out, int out_size, void* d_ws, size_t ws_size,
                              hipStream_t stream)
{
    const float* x         = (const float*)d_in[0];  // (2,1024,1024)
    const float* in_proj_w = (const float*)d_in[1];  // (4096,1024)
    const float* conv_w    = (const float*)d_in[2];  // (2048,4)
    const float* conv_b    = (const float*)d_in[3];  // (2048,)
    const float* x_proj_w  = (const float*)d_in[4];  // (96,2048)
    const float* dt_proj_w = (const float*)d_in[5];  // (2048,64)
    const float* dt_proj_b = (const float*)d_in[6];  // (2048,)
    const float* A_log     = (const float*)d_in[7];  // (2048,16)
    const float* Dv        = (const float*)d_in[8];  // (2048,)
    const float* out_proj_w= (const float*)d_in[9];  // (1024,2048)
    float* out = (float*)d_out;                      // (2,1024,1024)

    // workspace layout
    char* p = (char*)d_ws;
    size_t off = 0;
    auto alloc = [&](size_t bytes) { void* q = p + off; off = align256(off + bytes); return q; };
    __hip_bfloat16* x_bf    = (__hip_bfloat16*)alloc((size_t)NR * DM * 2);
    __hip_bfloat16* win_bf  = (__hip_bfloat16*)alloc((size_t)2 * DI * DM * 2);
    __hip_bfloat16* wxp_bf  = (__hip_bfloat16*)alloc((size_t)DBLW * DI * 2);
    __hip_bfloat16* wdt_bf  = (__hip_bfloat16*)alloc((size_t)DI * DR * 2);
    __hip_bfloat16* wout_bf = (__hip_bfloat16*)alloc((size_t)DM * DI * 2);
    float*          xz      = (float*)alloc((size_t)NR * 2 * DI * 4);
    float*          xc      = (float*)alloc((size_t)NR * DI * 4);
    __hip_bfloat16* xc_bf   = (__hip_bfloat16*)alloc((size_t)NR * DI * 2);
    float*          dbl     = (float*)alloc((size_t)NR * DBLW * 4);
    __hip_bfloat16* dtin_bf = (__hip_bfloat16*)alloc((size_t)NR * DR * 2);
    float*          dt      = (float*)alloc((size_t)NR * DI * 4);
    float*          ys      = (float*)alloc((size_t)NR * DI * 4);
    __hip_bfloat16* y_bf    = (__hip_bfloat16*)alloc((size_t)NR * DI * 2);
    (void)ws_size; (void)in_sizes; (void)n_in; (void)out_size;

    auto cvt = [&](const float* s, __hip_bfloat16* d, int n) {
        cvt_bf16_kernel<<<(n + 255) / 256, 256, 0, stream>>>(s, d, n);
    };
    cvt(x,          x_bf,    NR * DM);
    cvt(in_proj_w,  win_bf,  2 * DI * DM);
    cvt(x_proj_w,   wxp_bf,  DBLW * DI);
    cvt(dt_proj_w,  wdt_bf,  DI * DR);
    cvt(out_proj_w, wout_bf, DM * DI);

    // 1) xz = x @ in_proj_w^T        (2048x1024 x 4096x1024^T)
    {
        dim3 g(NR / BM, (2 * DI) / BN);
        gemm_bf16_kernel<0><<<g, 256, 0, stream>>>(x_bf, win_bf, nullptr, xz, NR, 2 * DI, DM);
    }
    // 2) causal conv + SiLU -> xc (f32 + bf16)
    conv_silu_kernel<<<(NR * DI + 255) / 256, 256, 0, stream>>>(xz, conv_w, conv_b, xc, xc_bf);
    // 3) dbl = xc @ x_proj_w^T       (N = 96, ragged tile)
    {
        dim3 g(NR / BM, (DBLW + BN - 1) / BN);
        gemm_bf16_kernel<0><<<g, 256, 0, stream>>>(xc_bf, wxp_bf, nullptr, dbl, NR, DBLW, DI);
    }
    // 4) dt_in slice -> bf16
    dtin_slice_kernel<<<(NR * DR + 255) / 256, 256, 0, stream>>>(dbl, dtin_bf);
    // 5) dt = softplus(dt_in @ dt_proj_w^T + b)
    {
        dim3 g(NR / BM, DI / BN);
        gemm_bf16_kernel<1><<<g, 256, 0, stream>>>(dtin_bf, wdt_bf, dt_proj_b, dt, NR, DI, DR);
    }
    // 6) selective scan (16 lanes per channel)
    scan_kernel<<<(B_ * DI * DS) / 256, 256, 0, stream>>>(dt, dbl, xc, A_log, ys);
    // 7) gate
    gate_kernel<<<(NR * DI + 255) / 256, 256, 0, stream>>>(ys, xc, Dv, xz, y_bf);
    // 8) out = y @ out_proj_w^T
    {
        dim3 g(NR / BM, DM / BN);
        gemm_bf16_kernel<0><<<g, 256, 0, stream>>>(y_bf, wout_bf, nullptr, out, NR, DM, DI);
    }
}
